// MoELayer_69226282877352
// MI455X (gfx1250) — matrix operand order repaired and checked by reading, not compiled, not run on hardware
//
#include <hip/hip_runtime.h>
#include <math.h>
#include <stdint.h>

// ---------------------------------------------------------------------------
// MoE layer for MI455X (gfx1250, wave32, WMMA + Tensor Data Mover).
// Sparse top-2 routing + bf16 WMMA expert FFN GEMMs with fp32 accumulation.
// Workspace use: ~35 MB (lists/partials < 1 MB + 32 MB bf16 H buffer).
// ---------------------------------------------------------------------------

typedef __attribute__((ext_vector_type(16))) __bf16 v16bf;
typedef __attribute__((ext_vector_type(8)))  __bf16 v8bf;
typedef __attribute__((ext_vector_type(4)))  __bf16 v4bf;
typedef __attribute__((ext_vector_type(8)))  float  floatx8;
typedef __attribute__((ext_vector_type(4)))  unsigned int v4u;
typedef __attribute__((ext_vector_type(4)))  int v4i;
typedef __attribute__((ext_vector_type(8)))  int v8i;

static constexpr int Bn = 4;
static constexpr int Sn = 2048;
static constexpr int Dn = 1024;
static constexpr int Fn = 4096;
static constexpr int En = 8;
static constexpr int Tn = Bn * Sn;            // 8192 tokens
static constexpr int SLOT_CHUNK = 4096;       // H rows per chunk launch
static constexpr int TM = 128, TN = 128, KS = 64;
static constexpr int LDT = KS + 8;            // LDS row stride (16B multiple)
static constexpr int GATE_BLKS = Tn / 8;      // 8 tokens per 256-thr block

__device__ __forceinline__ __bf16 f2bf(float f) {
    return (__bf16)f;                          // native v_cvt (RNE)
}

__device__ __forceinline__ float gelu_exact(float v) {
    return 0.5f * v * (1.0f + erff(v * 0.70710678118654752f));
}

// A-frag (16x32 bf16): lane L<16 -> row L, K {kb..kb+7} and {kb+16..kb+23},
// kb = 8*(L>>4).  Two 16B LDS reads.
__device__ __forceinline__ v16bf load_fragA(const __bf16* p0, int ld, int lane) {
    int r  = lane & 15;
    int kb = (lane >> 4) << 3;
    const __bf16* p = p0 + r * ld + kb;
    union { v16bf v; v8bf h[2]; } u;
    u.h[0] = *(const v8bf*)(p);
    u.h[1] = *(const v8bf*)(p + 16);
    return u.v;
}

// B-frag (32x16 bf16), tile staged transposed [n][k]: lane L -> column L&15,
// K {kb..kb+7} and {kb+16..kb+23}, kb = 8*(L>>4).  Two 16B LDS reads.
__device__ __forceinline__ v16bf load_fragB(const __bf16* p0, int ld, int lane) {
    int n  = lane & 15;
    int kb = (lane >> 4) << 3;
    const __bf16* p = p0 + n * ld + kb;
    union { v16bf v; v8bf h[2]; } u;
    u.h[0] = *(const v8bf*)(p);
    u.h[1] = *(const v8bf*)(p + 16);
    return u.v;
}

// ---------------------------------------------------------------------------
// Tensor Data Mover: 2-D bf16 tile load Global->LDS (D# per ISA 08 §8.3/8.4).
//   tile: TM(128) rows x KS(64) cols of a row-major [*, Fn] bf16 tensor.
//   rem_rows < TM rows are zero-filled by TDM OOB handling (tensor_dim1).
//   pad_enable: after every 32 DWORDs (one 64-elem row) insert 4 DWORDs,
//   reproducing the LDT=72-element LDS row stride.
// ---------------------------------------------------------------------------
__device__ __forceinline__ void tdm_load_tile(unsigned lds_off,
                                              const __bf16* gsrc,
                                              unsigned rem_rows) {
    unsigned long long ga = (unsigned long long)(uintptr_t)gsrc;
    v4u g0;
    g0[0] = 1u;                                        // count=1, load, user
    g0[1] = lds_off;                                   // LDS byte address
    g0[2] = (unsigned)(ga & 0xFFFFFFFFu);              // global_addr[31:0]
    g0[3] = (unsigned)((ga >> 32) & 0x01FFFFFFu)       // global_addr[56:32]
          | 0x80000000u;                               // type=2 ("image")
    v8i g1;
    g1[0] = (int)((1u << 16)                           // data_size=1 -> 2B
                | (1u << 20)                           // pad_enable
                | (4u << 22)                           // pad_interval: 32 DW
                | (3u << 25));                         // pad_amount: 4 DW
    g1[1] = (int)(((unsigned)Fn & 0xFFFFu) << 16);     // tensor_dim0[15:0]
    g1[2] = (int)(((unsigned)Fn >> 16)                 // tensor_dim0[31:16]
                | ((rem_rows & 0xFFFFu) << 16));       // tensor_dim1[15:0]
    g1[3] = (int)((rem_rows >> 16)                     // tensor_dim1[31:16]
                | ((unsigned)KS << 16));               // tile_dim0=64
    g1[4] = (int)TM;                                   // tile_dim1=128, dim2=0
    g1[5] = (int)Fn;                                   // dim0_stride[31:0]
    g1[6] = 0;                                         // stride hi / dim1_stride
    g1[7] = 0;
    v4i z4 = (v4i)0;
    v8i z8 = (v8i)0;
    __builtin_amdgcn_tensor_load_to_lds(g0, g1, z4, z4, z8, 0);
}

// ---------------------------------------------------------------------------
// K0: zero output (incl. loss slot) and expert counters.
// ---------------------------------------------------------------------------
__launch_bounds__(256)
__global__ void moe_zero(float* __restrict__ out, int n, int* __restrict__ cnt) {
    int i = blockIdx.x * blockDim.x + threadIdx.x;
    for (; i < n; i += gridDim.x * blockDim.x) out[i] = 0.0f;
    if (blockIdx.x == 0 && threadIdx.x < En) cnt[threadIdx.x] = 0;
}

// ---------------------------------------------------------------------------
// K1: gating. One wave per token: logits, softmax(+bias), top-2, renorm,
// append to per-expert lists; per-block loss partials in fixed order.
// ---------------------------------------------------------------------------
__launch_bounds__(256)
__global__ void moe_gate(const float* __restrict__ x,
                         const float* __restrict__ ew,
                         const float* __restrict__ gw,
                         const float* __restrict__ gb,
                         int* __restrict__ cnt,
                         int* __restrict__ tok,
                         float* __restrict__ wgt,
                         float* __restrict__ Fp,
                         float* __restrict__ Pp) {
    __shared__ float sP[8][En];
    __shared__ float sF[8][En];

    const int tid  = threadIdx.x;
    const int lane = tid & 31;
    const int w    = tid >> 5;
    const int t    = blockIdx.x * 8 + w;

    float acc[En];
#pragma unroll
    for (int e = 0; e < En; ++e) acc[e] = 0.0f;

    const float* xp = x + (size_t)t * Dn;
    for (int d = lane; d < Dn; d += 32) {
        float xv = xp[d];
        const float* g = gw + (size_t)d * En;
        float4 g0 = *(const float4*)(g);
        float4 g1 = *(const float4*)(g + 4);
        acc[0] += xv * g0.x; acc[1] += xv * g0.y;
        acc[2] += xv * g0.z; acc[3] += xv * g0.w;
        acc[4] += xv * g1.x; acc[5] += xv * g1.y;
        acc[6] += xv * g1.z; acc[7] += xv * g1.w;
    }
#pragma unroll
    for (int e = 0; e < En; ++e)
#pragma unroll
        for (int off = 16; off > 0; off >>= 1)
            acc[e] += __shfl_xor(acc[e], off, 32);

    if (lane == 0) {
        const int bidx = t / Sn;
        float l[En], p[En];
#pragma unroll
        for (int e = 0; e < En; ++e) l[e] = acc[e] + gb[e] + ew[bidx * En + e];
        float mx = l[0];
#pragma unroll
        for (int e = 1; e < En; ++e) mx = fmaxf(mx, l[e]);
        float sum = 0.0f;
#pragma unroll
        for (int e = 0; e < En; ++e) { p[e] = expf(l[e] - mx); sum += p[e]; }
        float inv = 1.0f / sum;
#pragma unroll
        for (int e = 0; e < En; ++e) p[e] *= inv;

        // top-1 (first max, jnp.argmax semantics)
        int i1 = 0; float p1 = p[0];
#pragma unroll
        for (int e = 1; e < En; ++e) if (p[e] > p1) { p1 = p[e]; i1 = e; }
        // top-2 (first occurrence of second max)
        int i2 = -1; float p2 = -1.0f;
#pragma unroll
        for (int e = 0; e < En; ++e)
            if (e != i1 && p[e] > p2) { p2 = p[e]; i2 = e; }

        float s2 = p1 + p2;
        float c1 = p1 / s2, c2 = p2 / s2;

        int pos1 = atomicAdd(&cnt[i1], 1);
        tok[i1 * Tn + pos1] = t; wgt[i1 * Tn + pos1] = c1;
        int pos2 = atomicAdd(&cnt[i2], 1);
        tok[i2 * Tn + pos2] = t; wgt[i2 * Tn + pos2] = c2;

#pragma unroll
        for (int e = 0; e < En; ++e) {
            sP[w][e] = p[e];
            sF[w][e] = (e == i1) ? 1.0f : 0.0f;
        }
    }
    __syncthreads();
    if (tid < En) {
        float fs = 0.0f, ps = 0.0f;
        for (int ww = 0; ww < 8; ++ww) { fs += sF[ww][tid]; ps += sP[ww][tid]; }
        Fp[blockIdx.x * En + tid] = fs;
        Pp[blockIdx.x * En + tid] = ps;
    }
}

// ---------------------------------------------------------------------------
// K2: load-balance loss (single block, fixed-order reduction -> deterministic)
// ---------------------------------------------------------------------------
__launch_bounds__(64)
__global__ void moe_loss(const float* __restrict__ Fp, const float* __restrict__ Pp,
                         float* __restrict__ loss_out) {
    __shared__ float fe[En], pe[En];
    int tid = threadIdx.x;
    if (tid < En) {
        float f = 0.0f, p = 0.0f;
        for (int b = 0; b < GATE_BLKS; ++b) {
            f += Fp[b * En + tid];
            p += Pp[b * En + tid];
        }
        fe[tid] = f; pe[tid] = p;
    }
    __syncthreads();
    if (tid == 0) {
        float s = 0.0f;
        for (int e = 0; e < En; ++e)
            s += (fe[e] / (float)Tn) * (pe[e] / (float)Tn);
        *loss_out = 0.01f * (float)En * s;
    }
}

// ---------------------------------------------------------------------------
// K3: pass 1 per expert chunk: H[s,f] = gelu(x[tok[s]] @ w1[e] + b1[e])
// 128x128 tile, K=1024 in 16 stages of 64, bf16 WMMA, fp32 accum.
// ---------------------------------------------------------------------------
__launch_bounds__(256)
__global__ void moe_ffn1(const float* __restrict__ x,
                         const float* __restrict__ w1,
                         const float* __restrict__ b1,
                         const int* __restrict__ cnt,
                         const int* __restrict__ tok,
                         __bf16* __restrict__ H,
                         int e, int slot_base) {
    int rows = cnt[e] - slot_base;
    if (rows > SLOT_CHUNK) rows = SLOT_CHUNK;
    const int m0 = blockIdx.x * TM;
    if (rows <= 0 || m0 >= rows) return;
    const int f0 = blockIdx.y * TN;

    __shared__ __bf16 As[TM][LDT];
    __shared__ __bf16 Bs[TN][LDT];   // transposed: [n][k]

    const int tid  = threadIdx.x;
    const int lane = tid & 31;
    const int w    = tid >> 5;
    const int mib  = (w >> 2) * 4;   // 4 row-subtiles per wave
    const int njb  = (w & 3) * 2;    // 2 col-subtiles per wave

    floatx8 c[4][2];
#pragma unroll
    for (int m = 0; m < 4; ++m)
#pragma unroll
        for (int j = 0; j < 2; ++j) c[m][j] = (floatx8)0.0f;

    const int* tokE = tok + e * Tn + slot_base;

    for (int k0 = 0; k0 < Dn; k0 += KS) {
        __syncthreads();
        {   // stage A: gathered token rows of x, fp32 -> bf16, 8B stores
            const int r  = tid >> 1;
            const int c0 = (tid & 1) * 32;
            const int s  = m0 + r;
            if (s < rows) {
                const float* xp = x + (size_t)tokE[s] * Dn + k0 + c0;
#pragma unroll
                for (int j = 0; j < 32; j += 4) {
                    float4 v = *(const float4*)(xp + j);
                    v4bf b;
                    b[0] = f2bf(v.x); b[1] = f2bf(v.y);
                    b[2] = f2bf(v.z); b[3] = f2bf(v.w);
                    *(v4bf*)(&As[r][c0 + j]) = b;
                }
            } else {
                v4bf zb4 = (v4bf)f2bf(0.0f);
#pragma unroll
                for (int j = 0; j < 32; j += 4)
                    *(v4bf*)(&As[r][c0 + j]) = zb4;
            }
        }
        {   // stage B: w1[e][k0+kk][f0+n], store transposed [n][kk]
            const int kk = tid >> 2;
            const int n0 = (tid & 3) * 32;
            const float* wp = w1 + ((size_t)e * Dn + (k0 + kk)) * Fn + f0 + n0;
#pragma unroll
            for (int j = 0; j < 32; j += 4) {
                float4 v = *(const float4*)(wp + j);
                Bs[n0 + j + 0][kk] = f2bf(v.x);
                Bs[n0 + j + 1][kk] = f2bf(v.y);
                Bs[n0 + j + 2][kk] = f2bf(v.z);
                Bs[n0 + j + 3][kk] = f2bf(v.w);
            }
        }
        __syncthreads();
#pragma unroll
        for (int ks = 0; ks < KS; ks += 32) {
            v16bf a[4], bfr[2];
#pragma unroll
            for (int m = 0; m < 4; ++m)
                a[m] = load_fragA(&As[(mib + m) * 16][ks], LDT, lane);
#pragma unroll
            for (int j = 0; j < 2; ++j)
                bfr[j] = load_fragB(&Bs[(njb + j) * 16][ks], LDT, lane);
#pragma unroll
            for (int m = 0; m < 4; ++m)
#pragma unroll
                for (int j = 0; j < 2; ++j)
                    c[m][j] = __builtin_amdgcn_wmma_f32_16x16x32_bf16(
                        false, a[m], false, bfr[j], (short)0, c[m][j],
                        false, false);
        }
    }

    // epilogue: + b1, exact gelu, store bf16 H (chunk-local rows)
    const int cn = lane & 15;
    const int rb = (lane >> 4) * 8;
#pragma unroll
    for (int m = 0; m < 4; ++m) {
#pragma unroll
        for (int j = 0; j < 2; ++j) {
            const int fcol = f0 + (njb + j) * 16 + cn;
            const float bb = b1[e * Fn + fcol];
#pragma unroll
            for (int i = 0; i < 8; ++i) {
                const int s = m0 + (mib + m) * 16 + rb + i;
                if (s < rows)
                    H[(size_t)s * Fn + fcol] = f2bf(gelu_exact(c[m][j][i] + bb));
            }
        }
    }
}

// ---------------------------------------------------------------------------
// K4: pass 2 per expert chunk: out[tok[s],:] += wgt[s]*(H[s,:] @ w2[e] + b2[e])
// A tile staged by the Tensor Data Mover (async, zero-fills ragged tails);
// exactly 2 commutative fp32 atomic adds per out element -> deterministic.
// ---------------------------------------------------------------------------
__launch_bounds__(256)
__global__ void moe_ffn2(const __bf16* __restrict__ H,
                         const float* __restrict__ w2,
                         const float* __restrict__ b2,
                         const int* __restrict__ cnt,
                         const int* __restrict__ tok,
                         const float* __restrict__ wgt,
                         float* __restrict__ out,
                         int e, int slot_base) {
    int rows = cnt[e] - slot_base;
    if (rows > SLOT_CHUNK) rows = SLOT_CHUNK;
    const int m0 = blockIdx.x * TM;
    if (rows <= 0 || m0 >= rows) return;
    const int d0 = blockIdx.y * TN;
    const unsigned rem_rows = (unsigned)(rows - m0);

    __shared__ __bf16 As[TM][LDT];
    __shared__ __bf16 Bs[TN][LDT];   // transposed: [n][k]

    const int tid  = threadIdx.x;
    const int lane = tid & 31;
    const int w    = tid >> 5;
    const int mib  = (w >> 2) * 4;
    const int njb  = (w & 3) * 2;

    const unsigned as_lds = (unsigned)(uintptr_t)(&As[0][0]);

    floatx8 c[4][2];
#pragma unroll
    for (int m = 0; m < 4; ++m)
#pragma unroll
        for (int j = 0; j < 2; ++j) c[m][j] = (floatx8)0.0f;

    for (int k0 = 0; k0 < Fn; k0 += KS) {
        __syncthreads();
        if (w == 0) {
            // async DMA: H[m0 .. m0+127][k0 .. k0+63] -> As (padded stride)
            tdm_load_tile(as_lds, H + (size_t)m0 * Fn + k0, rem_rows);
        }
        {   // stage B: w2[e][k0+kk][d0+n] transposed (overlaps the TDM)
            const int kk = tid >> 2;
            const int n0 = (tid & 3) * 32;
            const float* wp = w2 + ((size_t)e * Fn + (k0 + kk)) * Dn + d0 + n0;
#pragma unroll
            for (int j = 0; j < 32; j += 4) {
                float4 v = *(const float4*)(wp + j);
                Bs[n0 + j + 0][kk] = f2bf(v.x);
                Bs[n0 + j + 1][kk] = f2bf(v.y);
                Bs[n0 + j + 2][kk] = f2bf(v.z);
                Bs[n0 + j + 3][kk] = f2bf(v.w);
            }
        }
        if (w == 0) __builtin_amdgcn_s_wait_tensorcnt(0);
        __syncthreads();
#pragma unroll
        for (int ks = 0; ks < KS; ks += 32) {
            v16bf a[4], bfr[2];
#pragma unroll
            for (int m = 0; m < 4; ++m)
                a[m] = load_fragA(&As[(mib + m) * 16][ks], LDT, lane);
#pragma unroll
            for (int j = 0; j < 2; ++j)
                bfr[j] = load_fragB(&Bs[(njb + j) * 16][ks], LDT, lane);
#pragma unroll
            for (int m = 0; m < 4; ++m)
#pragma unroll
                for (int j = 0; j < 2; ++j)
                    c[m][j] = __builtin_amdgcn_wmma_f32_16x16x32_bf16(
                        false, a[m], false, bfr[j], (short)0, c[m][j],
                        false, false);
        }
    }

    // epilogue: scale by routing weight, scatter-add into out
    const int cn = lane & 15;
    const int rb = (lane >> 4) * 8;
    const int* tokE = tok + e * Tn + slot_base;
    const float* wgtE = wgt + e * Tn + slot_base;
#pragma unroll
    for (int m = 0; m < 4; ++m) {
#pragma unroll
        for (int j = 0; j < 2; ++j) {
            const int dcol = d0 + (njb + j) * 16 + cn;
            const float bb = b2[e * Dn + dcol];
#pragma unroll
            for (int i = 0; i < 8; ++i) {
                const int s = m0 + (mib + m) * 16 + rb + i;
                if (s < rows) {
                    const int   t = tokE[s];
                    const float g = wgtE[s];
                    atomicAdd(&out[(size_t)t * Dn + dcol], g * (c[m][j][i] + bb));
                }
            }
        }
    }
}

// ---------------------------------------------------------------------------
extern "C" void kernel_launch(void* const* d_in, const int* in_sizes, int n_in,
                              void* d_out, int out_size, void* d_ws, size_t ws_size,
                              hipStream_t stream) {
    (void)in_sizes; (void)n_in; (void)out_size; (void)ws_size;

    const float* x  = (const float*)d_in[0];
    const float* ew = (const float*)d_in[1];
    const float* gw = (const float*)d_in[2];
    const float* gb = (const float*)d_in[3];
    const float* w1 = (const float*)d_in[4];
    const float* b1 = (const float*)d_in[5];
    const float* w2 = (const float*)d_in[6];
    const float* b2 = (const float*)d_in[7];
    float* out = (float*)d_out;

    // workspace layout (bytes): [cnt 8i | tok E*T i | wgt E*T f | Fp | Pp | H]
    char* ws = (char*)d_ws;
    int*    cnt = (int*)(ws + 0);
    int*    tok = (int*)(ws + 256);
    float*  wgt = (float*)(ws + 256 + 4 * En * Tn);
    float*  Fp  = (float*)(ws + 256 + 8 * En * Tn);
    float*  Pp  = (float*)(ws + 256 + 8 * En * Tn + 4 * En * GATE_BLKS);
    __bf16* H   = (__bf16*)(ws + (1 << 20));   // 32 MB: 4096 x 4096 bf16

    moe_zero<<<2048, 256, 0, stream>>>(out, Tn * Dn + 1, cnt);
    moe_gate<<<GATE_BLKS, 256, 0, stream>>>(x, ew, gw, gb, cnt, tok, wgt, Fp, Pp);
    moe_loss<<<1, 64, 0, stream>>>(Fp, Pp, out + (size_t)Tn * Dn);

    for (int e = 0; e < En; ++e) {
        for (int cidx = 0; cidx < Tn / SLOT_CHUNK; ++cidx) {
            const int sb = cidx * SLOT_CHUNK;
            moe_ffn1<<<dim3(SLOT_CHUNK / TM, Fn / TN), 256, 0, stream>>>(
                x, w1, b1, cnt, tok, H, e, sb);
            moe_ffn2<<<dim3(SLOT_CHUNK / TM, Dn / TN), 256, 0, stream>>>(
                H, w2, b2, cnt, tok, wgt, out, e, sb);
        }
    }
}
